// NodeDegreeEmbeddingNetwork_65876208386286
// MI455X (gfx1250) — matrix operand order repaired and checked by reading, not compiled, not run on hardware
//
#include <hip/hip_runtime.h>

typedef __bf16 bf16;
typedef __attribute__((ext_vector_type(16))) __bf16 v16bf;
typedef __attribute__((ext_vector_type(8)))  __bf16 bf16x8;
typedef __attribute__((ext_vector_type(8)))  float   v8f;
typedef __attribute__((ext_vector_type(4)))  float   f32x4;

__device__ __forceinline__ v8f wmma_bf16(v16bf a, v16bf b, v8f c) {
  // D = A(16x32) * B(32x16) + C, f32 accumulate
  return __builtin_amdgcn_wmma_f32_16x16x32_bf16(false, a, false, b, (short)0, c,
                                                 false, false);
}

// ---------------------------------------------------------------- zero output
__global__ void zero_kernel(float* __restrict__ p, int n) {
  int i = blockIdx.x * blockDim.x + threadIdx.x;
  int stride = gridDim.x * blockDim.x;
  for (; i < n; i += stride) p[i] = 0.f;
}

// ------------------------------------------------- prep1: P = ns*proj_w, W1t
// ns[c] = sum_r exp_w[r][c] + exp_b[c];  P[c][d] = ns[c]*proj_w[c][d]
// W1t[n][k] = bf16(rad_w1[k][n])   (k-contiguous, WMMA-B-ready)
__global__ __launch_bounds__(256) void prep1_kernel(
    const float* __restrict__ exp_w, const float* __restrict__ exp_b,
    const float* __restrict__ proj_w, const float* __restrict__ rad_w1,
    float* __restrict__ P, bf16* __restrict__ W1t) {
  __shared__ float ns[64];
  int tid = threadIdx.x;
  if (tid < 64) {
    float s = exp_b[tid];
    for (int r = 0; r < 64; ++r) s += exp_w[r * 64 + tid];
    ns[tid] = s;
  }
  __syncthreads();
  for (int idx = tid; idx < 4096; idx += 256) {
    int c = idx >> 6, d = idx & 63;
    P[idx] = ns[c] * proj_w[idx];
    W1t[d * 64 + c] = (bf16)rad_w1[idx];
  }
}

// ---------------------- prep2: W2Pt[j][c] = sum_k rad_w2[c][l*64+k]*P[k][d]
// (j = l*64+d), offP[j] = sum_k rad_offset[l*64+k]*P[k][d]
__global__ __launch_bounds__(64) void prep2_kernel(
    const float* __restrict__ rad_w2, const float* __restrict__ rad_offset,
    const float* __restrict__ P, bf16* __restrict__ W2Pt,
    float* __restrict__ offP) {
  int j = blockIdx.x;  // 0..191
  int l = j / 64, d = j & 63;
  int c = threadIdx.x;
  __shared__ float pcol[64];
  pcol[c] = P[c * 64 + d];
  __syncthreads();
  float acc = 0.f;
  for (int k = 0; k < 64; ++k) acc += rad_w2[c * 192 + l * 64 + k] * pcol[k];
  W2Pt[j * 64 + c] = (bf16)acc;
  if (c == 0) {
    float o = 0.f;
    for (int k = 0; k < 64; ++k) o += rad_offset[l * 64 + k] * pcol[k];
    offP[j] = o;
  }
}

// ------------------------------------------------------------- main per-edge
__global__ __launch_bounds__(32) void edge_kernel(
    const float* __restrict__ edge_scalars, const float* __restrict__ edge_attr,
    const int* __restrict__ edge_dst, const float* __restrict__ rad_b1,
    const float* __restrict__ ln_g, const float* __restrict__ ln_b,
    const float* __restrict__ proj_b, const bf16* __restrict__ W1t,
    const bf16* __restrict__ W2Pt, const float* __restrict__ offP,
    float* __restrict__ out, int E) {
  const int lane = threadIdx.x;
  const int m = lane & 15;
  const int half = lane >> 4;
  const long eb = (long)blockIdx.x * 16;
  int nvalid = (int)((E - eb < 16) ? (E - eb) : 16);

  __shared__ bf16 hsh[16 * 64];
  __shared__ float ea_sh[16 * 9];
  __shared__ int dst_sh[16];

  // stage epilogue data (edge_attr rows are contiguous: 9 per edge)
  for (int i = lane; i < nvalid * 9; i += 32) ea_sh[i] = edge_attr[eb * 9 + i];
  if (lane < 16) dst_sh[lane] = (lane < nvalid) ? edge_dst[eb + lane] : 0;

  // ---- A tiles (16 edges x 64 ch) f32 -> bf16, WMMA A-layout:
  // lane(m,half): elem j -> k = t*32 + half*16 + j
  long em = eb + ((m < nvalid) ? m : (nvalid - 1));
  const float* esrow = edge_scalars + em * 64;
  v16bf a[2];
#pragma unroll
  for (int t = 0; t < 2; ++t) {
    const float* p0 = esrow + t * 32 + half * 16;
    f32x4 c0 = *(const f32x4*)(p0);
    f32x4 c1 = *(const f32x4*)(p0 + 4);
    f32x4 c2 = *(const f32x4*)(p0 + 8);
    f32x4 c3 = *(const f32x4*)(p0 + 12);
#pragma unroll
    for (int i = 0; i < 4; ++i) {
      a[t][i]      = (bf16)c0[i];
      a[t][4 + i]  = (bf16)c1[i];
      a[t][8 + i]  = (bf16)c2[i];
      a[t][12 + i] = (bf16)c3[i];
    }
  }

  // ---- GEMM1: t = A @ rad_w1 + b1   (4 n-tiles x 2 k-steps)
  // B-layout: lane(n=m, half): elem j -> k = t*32 + half*16 + j  (k-contig)
  v8f tacc[4];
#pragma unroll
  for (int nt = 0; nt < 4; ++nt) {
    float bias = rad_b1[nt * 16 + m];
    v8f c;
#pragma unroll
    for (int r = 0; r < 8; ++r) c[r] = bias;
    const bf16* bp = W1t + (nt * 16 + m) * 64 + half * 16;
    v16bf b0 = *(const v16bf*)(bp);
    v16bf b1 = *(const v16bf*)(bp + 32);
    c = wmma_bf16(a[0], b0, c);
    c = wmma_bf16(a[1], b1, c);
    tacc[nt] = c;
  }

  // ---- LayerNorm over 64 channels per row + SiLU (rows live in 16-lane halves)
  float g4[4], b4[4];
#pragma unroll
  for (int nt = 0; nt < 4; ++nt) {
    g4[nt] = ln_g[nt * 16 + m];
    b4[nt] = ln_b[nt * 16 + m];
  }
#pragma unroll
  for (int r = 0; r < 8; ++r) {
    float s = 0.f, q = 0.f;
#pragma unroll
    for (int nt = 0; nt < 4; ++nt) {
      float x = tacc[nt][r];
      s += x;
      q += x * x;
    }
#pragma unroll
    for (int msk = 1; msk <= 8; msk <<= 1) {
      s += __shfl_xor(s, msk, 32);
      q += __shfl_xor(q, msk, 32);
    }
    float mu = s * 0.015625f;
    float var = q * 0.015625f - mu * mu;
    float inv = rsqrtf(var + 1e-5f);
#pragma unroll
    for (int nt = 0; nt < 4; ++nt) {
      float x = (tacc[nt][r] - mu) * inv * g4[nt] + b4[nt];
      tacc[nt][r] = x / (1.f + __expf(-x));  // silu
    }
  }

  // ---- bounce h through LDS (C-layout -> A-layout), as bf16 row-major [m][64]
#pragma unroll
  for (int nt = 0; nt < 4; ++nt)
#pragma unroll
    for (int r = 0; r < 8; ++r)
      hsh[(r + 8 * half) * 64 + nt * 16 + m] = (bf16)tacc[nt][r];
  __syncthreads();  // single-wave WG: S_NOP at runtime, compile-time fence

  v16bf ha[2];
#pragma unroll
  for (int t = 0; t < 2; ++t) {
    const bf16* hp = hsh + m * 64 + t * 32 + half * 16;
    bf16x8 lo = *(const bf16x8*)(hp);
    bf16x8 hi = *(const bf16x8*)(hp + 8);
#pragma unroll
    for (int i = 0; i < 8; ++i) {
      ha[t][i] = lo[i];
      ha[t][8 + i] = hi[i];
    }
  }

  // ---- fused GEMM2: M(16x192) = h @ W2P + offP   (12 n-tiles x 2 k-steps)
  v8f macc[12];
#pragma unroll
  for (int jt = 0; jt < 12; ++jt) {
    float o = offP[jt * 16 + m];
    v8f c;
#pragma unroll
    for (int r = 0; r < 8; ++r) c[r] = o;
    const bf16* bp = W2Pt + (jt * 16 + m) * 64 + half * 16;
    v16bf b0 = *(const v16bf*)(bp);
    v16bf b1 = *(const v16bf*)(bp + 32);
    c = wmma_bf16(ha[0], b0, c);
    c = wmma_bf16(ha[1], b1, c);
    macc[jt] = c;
  }

  // ---- epilogue: out[dst, lm, d] += 0.25*(edge_attr[e,lm]*M[e,l(lm),d] (+pb))
  float pb4[4];
#pragma unroll
  for (int nt = 0; nt < 4; ++nt) pb4[nt] = proj_b[nt * 16 + m];
  constexpr int lut[9] = {0, 1, 1, 1, 2, 2, 2, 2, 2};
#pragma unroll
  for (int lm = 0; lm < 9; ++lm) {
    const int l = lut[lm];
#pragma unroll
    for (int nt = 0; nt < 4; ++nt) {
      const int d = nt * 16 + m;
      v8f acc = macc[l * 4 + nt];
#pragma unroll
      for (int r = 0; r < 8; ++r) {
        int row = r + 8 * half;
        if (row < nvalid) {
          float v = acc[r] * ea_sh[row * 9 + lm];
          if (lm == 0) v += pb4[nt];
          unsafeAtomicAdd(out + (long)dst_sh[row] * 576 + lm * 64 + d,
                          0.25f * v);
        }
      }
    }
  }
}

// ---------------------------------------------------------------------------
extern "C" void kernel_launch(void* const* d_in, const int* in_sizes, int n_in,
                              void* d_out, int out_size, void* d_ws,
                              size_t ws_size, hipStream_t stream) {
  // setup_inputs order:
  // 0 node_input (unused), 1 edge_attr, 2 edge_scalars, 3 edge_src (unused),
  // 4 edge_dst, 5 exp_w, 6 exp_b, 7 rad_w1, 8 rad_b1, 9 rad_ln_g, 10 rad_ln_b,
  // 11 rad_w2, 12 rad_offset, 13 proj_w, 14 proj_b
  const float* edge_attr    = (const float*)d_in[1];
  const float* edge_scalars = (const float*)d_in[2];
  const int*   edge_dst     = (const int*)d_in[4];
  const float* exp_w   = (const float*)d_in[5];
  const float* exp_b   = (const float*)d_in[6];
  const float* rad_w1  = (const float*)d_in[7];
  const float* rad_b1  = (const float*)d_in[8];
  const float* ln_g    = (const float*)d_in[9];
  const float* ln_b    = (const float*)d_in[10];
  const float* rad_w2  = (const float*)d_in[11];
  const float* rad_off = (const float*)d_in[12];
  const float* proj_w  = (const float*)d_in[13];
  const float* proj_b  = (const float*)d_in[14];
  float* out = (float*)d_out;
  int E = in_sizes[4];

  // workspace layout (needs ~50 KB)
  char* ws = (char*)d_ws;
  float* P    = (float*)(ws);          // 4096 f32  = 16384 B
  float* offP = (float*)(ws + 16384);  // 192 f32   = 768 B
  bf16*  W1t  = (bf16*)(ws + 17408);   // 4096 bf16 = 8192 B
  bf16*  W2Pt = (bf16*)(ws + 25600);   // 12288 bf16 = 24576 B

  hipLaunchKernelGGL(zero_kernel, dim3(2048), dim3(256), 0, stream, out,
                     out_size);
  hipLaunchKernelGGL(prep1_kernel, dim3(1), dim3(256), 0, stream, exp_w, exp_b,
                     proj_w, rad_w1, P, W1t);
  hipLaunchKernelGGL(prep2_kernel, dim3(192), dim3(64), 0, stream, rad_w2,
                     rad_off, P, W2Pt, offP);
  int ntiles = (E + 15) / 16;
  hipLaunchKernelGGL(edge_kernel, dim3(ntiles), dim3(32), 0, stream,
                     edge_scalars, edge_attr, edge_dst, rad_b1, ln_g, ln_b,
                     proj_b, W1t, W2Pt, offP, out, E);
}
